// GraphConvolution_78709570667185
// MI455X (gfx1250) — matrix operand order repaired and checked by reading, not compiled, not run on hardware
//
#include <hip/hip_runtime.h>
#include <hip/hip_bf16.h>

typedef __attribute__((ext_vector_type(16))) __bf16 v16bf;
typedef __attribute__((ext_vector_type(8)))  __bf16 v8bf;
typedef __attribute__((ext_vector_type(8)))  float  v8f;

#define DCOL 256
#define NT_PER_WAVE 4   // N-tiles (16 cols each) per wave -> 16x64 strip

// ---------------------------------------------------------------------------
// out[i] = b[i % 256]   (bias pre-load; atomics accumulate on top)
// ---------------------------------------------------------------------------
__global__ void gcn_init_out(float* __restrict__ out, const float* __restrict__ b, int total) {
    int i = blockIdx.x * blockDim.x + threadIdx.x;
    if (i < total) out[i] = b[i & (DCOL - 1)];
}

// ---------------------------------------------------------------------------
// Pack W (256x256 f32 row-major) -> bf16 in WMMA B-fragment order.
// Fragment (kb, tn): 32 lanes x 16 bf16, lane l element i holds
//   W[kb*32 + (i>>3)*16 + (l>>4)*8 + (i&7)][tn*16 + (l&15)]
// Packed index: ((kb*16 + tn)*32 + l)*16 + i   (65536 elements = 128 KB)
// ---------------------------------------------------------------------------
__global__ void gcn_pack_w(const float* __restrict__ W, __bf16* __restrict__ Wp) {
    int p = blockIdx.x * blockDim.x + threadIdx.x;   // 0 .. 65535
    if (p >= 8 * 16 * 32 * 16) return;
    int i    = p & 15;
    int l    = (p >> 4) & 31;
    int tn   = (p >> 9) & 15;
    int kb   = p >> 13;
    int k    = kb * 32 + (i >> 3) * 16 + (l >> 4) * 8 + (i & 7);
    int n    = tn * 16 + (l & 15);
    Wp[p] = (__bf16)W[k * DCOL + n];
}

// ---------------------------------------------------------------------------
// H = X (M x 256, f32) * W (256 x 256, packed bf16), stored as bf16.
// One wave computes a 16x64 strip: 4 accumulators, 8 K-steps, 32 WMMAs.
// A fragments loaded/converted once per K-step, reused by 4 WMMAs.
// ---------------------------------------------------------------------------
__global__ void gcn_gemm_wmma(const float* __restrict__ X,
                              const __bf16* __restrict__ Wp,
                              __bf16* __restrict__ H,
                              int m_tiles) {
    const int lane    = threadIdx.x & 31;
    const int wave    = threadIdx.x >> 5;
    const int strip   = blockIdx.x * (blockDim.x >> 5) + wave;
    const int strips  = DCOL / (16 * NT_PER_WAVE);     // 4 strips of 64 cols
    if (strip >= m_tiles * strips) return;

    const int tm   = strip / strips;                   // M tile (16 rows)
    const int ng   = strip % strips;                   // which 64-col group
    const int m    = lane & 15;
    const int half = lane >> 4;

    v8f c0 = {}, c1 = {}, c2 = {}, c3 = {};
    const float* xrow = X + (size_t)(tm * 16 + m) * DCOL;

    #pragma unroll
    for (int kb = 0; kb < 8; ++kb) {
        // A fragment (16x32 bf16): lane holds row m.
        //   elems 0..7  -> k = kb*32 + half*8 + j
        //   elems 8..15 -> k = kb*32 + 16 + half*8 + j
        v16bf a;
        const float* xa = xrow + kb * 32 + half * 8;
        #pragma unroll
        for (int j = 0; j < 8; ++j) {
            a[j]     = (__bf16)xa[j];
            a[8 + j] = (__bf16)xa[16 + j];
        }
        // B fragments: contiguous 32-byte vector load per lane (pre-packed)
        const __bf16* wb = Wp + (((size_t)kb * 16 + ng * NT_PER_WAVE) * 32 + lane) * 16;
        v16bf b0 = *(const v16bf*)(wb);
        v16bf b1 = *(const v16bf*)(wb + 32 * 16);
        v16bf b2 = *(const v16bf*)(wb + 2 * 32 * 16);
        v16bf b3 = *(const v16bf*)(wb + 3 * 32 * 16);

        c0 = __builtin_amdgcn_wmma_f32_16x16x32_bf16(false, a, false, b0, (short)0, c0, false, false);
        c1 = __builtin_amdgcn_wmma_f32_16x16x32_bf16(false, a, false, b1, (short)0, c1, false, false);
        c2 = __builtin_amdgcn_wmma_f32_16x16x32_bf16(false, a, false, b2, (short)0, c2, false, false);
        c3 = __builtin_amdgcn_wmma_f32_16x16x32_bf16(false, a, false, b3, (short)0, c3, false, false);
    }

    // D layout: elem r -> row tm*16 + r + 8*half, col tn*16 + m
    __bf16* hp = H + (size_t)(tm * 16 + 8 * half) * DCOL + ng * NT_PER_WAVE * 16 + m;
    #pragma unroll
    for (int r = 0; r < 8; ++r) {
        hp[r * DCOL]          = (__bf16)c0[r];
        hp[r * DCOL + 16]     = (__bf16)c1[r];
        hp[r * DCOL + 32]     = (__bf16)c2[r];
        hp[r * DCOL + 48]     = (__bf16)c3[r];
    }
}

// ---------------------------------------------------------------------------
// SpMM scatter: one wave per edge. lane covers columns [lane*8, lane*8+8).
// out[dst] += val * H[src]  via global_atomic_add_f32 (no return).
// ---------------------------------------------------------------------------
__global__ void gcn_spmm_scatter(const int* __restrict__ src,
                                 const int* __restrict__ dst,
                                 const float* __restrict__ val,
                                 const __bf16* __restrict__ H,
                                 float* __restrict__ out, int E) {
    int gwave = (blockIdx.x * blockDim.x + threadIdx.x) >> 5;
    int lane  = threadIdx.x & 31;
    if (gwave >= E) return;

    // edge streams are consumed linearly from HBM: prefetch ahead
    if (lane == 0 && gwave + 4096 < E) {
        __builtin_prefetch(src + gwave + 4096, 0, 1);
        __builtin_prefetch(dst + gwave + 4096, 0, 1);
        __builtin_prefetch(val + gwave + 4096, 0, 1);
    }

    int   s = src[gwave];
    int   d = dst[gwave];
    float v = val[gwave];

    // 16-byte aligned vector gather of 8 bf16 (L2-resident)
    v8bf h = *(const v8bf*)(H + (size_t)s * DCOL + lane * 8);
    float* op = out + (size_t)d * DCOL + lane * 8;
    #pragma unroll
    for (int j = 0; j < 8; ++j) {
        unsafeAtomicAdd(op + j, (float)h[j] * v);
    }
}

// ---------------------------------------------------------------------------
// In-place ReLU
// ---------------------------------------------------------------------------
__global__ void gcn_relu(float* __restrict__ out, int total) {
    int i = blockIdx.x * blockDim.x + threadIdx.x;
    if (i < total) out[i] = fmaxf(out[i], 0.0f);
}

// ---------------------------------------------------------------------------
// inputs: 0:x 1:W0 2:W1 3:b 4:src0 5:dst0 6:val0 7:src1 8:dst1 9:val1
// ---------------------------------------------------------------------------
extern "C" void kernel_launch(void* const* d_in, const int* in_sizes, int n_in,
                              void* d_out, int out_size, void* d_ws, size_t ws_size,
                              hipStream_t stream) {
    const float* x    = (const float*)d_in[0];
    const float* W0   = (const float*)d_in[1];
    const float* W1   = (const float*)d_in[2];
    const float* b    = (const float*)d_in[3];
    const int*   src0 = (const int*)d_in[4];
    const int*   dst0 = (const int*)d_in[5];
    const float* val0 = (const float*)d_in[6];
    const int*   src1 = (const int*)d_in[7];
    const int*   dst1 = (const int*)d_in[8];
    const float* val1 = (const float*)d_in[9];
    float* out = (float*)d_out;

    const int N = in_sizes[0] / DCOL;      // 50000
    const int E = in_sizes[4];             // 1600000
    const int total = N * DCOL;

    // workspace layout: Wp0 | Wp1 | H0 | H1   (all bf16)
    __bf16* Wp0 = (__bf16*)d_ws;
    __bf16* Wp1 = Wp0 + DCOL * DCOL;
    __bf16* H0  = Wp1 + DCOL * DCOL;
    __bf16* H1  = H0 + (size_t)N * DCOL;

    // 1) out = bias
    gcn_init_out<<<(total + 255) / 256, 256, 0, stream>>>(out, b, total);

    // 2) pack weights into bf16 B-fragment layout (128 KB each)
    gcn_pack_w<<<(DCOL * DCOL + 255) / 256, 256, 0, stream>>>(W0, Wp0);
    gcn_pack_w<<<(DCOL * DCOL + 255) / 256, 256, 0, stream>>>(W1, Wp1);

    // 3) H0 = x@W0, H1 = x@W1 (bf16 WMMA, f32 accumulate, 16x64 strips)
    const int m_tiles = N / 16;                          // 3125
    const int strips  = m_tiles * (DCOL / (16 * NT_PER_WAVE)); // 12500 waves
    const int gblocks = (strips + 7) / 8;                // 8 waves / block
    gcn_gemm_wmma<<<gblocks, 256, 0, stream>>>(x, Wp0, H0, m_tiles);
    gcn_gemm_wmma<<<gblocks, 256, 0, stream>>>(x, Wp1, H1, m_tiles);

    // 4) edge scatter with f32 atomics (L2-resident h and out)
    const int sblocks = (E + 7) / 8;                     // 8 edges (waves) / block
    gcn_spmm_scatter<<<sblocks, 256, 0, stream>>>(src0, dst0, val0, H0, out, E);
    gcn_spmm_scatter<<<sblocks, 256, 0, stream>>>(src1, dst1, val1, H1, out, E);

    // 5) ReLU
    gcn_relu<<<(total + 255) / 256, 256, 0, stream>>>(out, total);
}
